// StationFlowGAT_75488345194668
// MI455X (gfx1250) — matrix operand order repaired and checked by reading, not compiled, not run on hardware
//
#include <hip/hip_runtime.h>
#include <math.h>

// ---------------------------------------------------------------------------
// GATv2 x4 layers for MI455X (gfx1250, wave32, WMMA, LDS-staged weights)
// ---------------------------------------------------------------------------

typedef __attribute__((ext_vector_type(16))) _Float16 v16h;
typedef __attribute__((ext_vector_type(8)))  _Float16 v8h;
typedef __attribute__((ext_vector_type(8)))  float    v8f;

// ---------------------------------------------------------------------------
// fp32 -> fp16 conversion (hoisted out of the GEMM hot loop).
// n8 = element_count / 8; element_count % 8 == 0 for all uses here.
// ---------------------------------------------------------------------------
__global__ void gat_cvt_f16(const float* __restrict__ s, _Float16* __restrict__ d,
                            long long n8) {
  long long i = (long long)blockIdx.x * blockDim.x + threadIdx.x;
  if (i >= n8) return;
  const float4* sp = (const float4*)s;
  float4 a = sp[2 * i];
  float4 b = sp[2 * i + 1];
  v8h o;
  o[0] = (_Float16)a.x; o[1] = (_Float16)a.y; o[2] = (_Float16)a.z; o[3] = (_Float16)a.w;
  o[4] = (_Float16)b.x; o[5] = (_Float16)b.y; o[6] = (_Float16)b.z; o[7] = (_Float16)b.w;
  ((v8h*)d)[i] = o;
}

// ---------------------------------------------------------------------------
// GEMM: Y[n,o] = sum_k X16[n,k] * W16[o,k] + bias[o]       (fp16 in, fp32 out)
// - whole W (O x K fp16, <=128KB) staged in LDS by the 256-thread block
// - 8 waves per block, one 16-row M-tile per wave
// - each wave accumulates OC 16x16 output tiles per A fragment (OC wmma / K step)
// Nrows % 16 == 0, K % 32 == 0, (O/16) % OC == 0.
// ---------------------------------------------------------------------------
template <int OC>
__global__ __launch_bounds__(256) void gat_gemm_wmma(
    const _Float16* __restrict__ X16, const _Float16* __restrict__ W16,
    const float* __restrict__ bias, float* __restrict__ Y,
    int Nrows, int K, int O) {
  extern __shared__ _Float16 sW[];   // O*K fp16

  // ---- cooperative stage of W into LDS (ds_store_b128) ----
  {
    const uint4* g = (const uint4*)W16;
    uint4* s = (uint4*)sW;
    const int nchunks = (O * K) >> 3;          // 16B chunks
    for (int i = threadIdx.x; i < nchunks; i += 256) s[i] = g[i];
  }
  __syncthreads();

  const int tilesM = Nrows >> 4;
  const int waveId = threadIdx.x >> 5;
  const int tm = blockIdx.x * 8 + waveId;      // uniform per wave
  if (tm >= tilesM) return;                    // EXEC all-1s inside live waves

  const int lane = threadIdx.x & 31;
  const int half = lane >> 4;                  // 0: lanes 0-15, 1: lanes 16-31
  const int l15 = lane & 15;

  const _Float16* __restrict__ arowp = X16 + (size_t)(tm * 16 + l15) * K;

  const int tilesO = O >> 4;
  for (int oc = 0; oc < tilesO; oc += OC) {
    v8f acc[OC];
    #pragma unroll
    for (int t = 0; t < OC; ++t) acc[t] = (v8f){};

    // per-lane LDS row base for each of the OC B columns
    const _Float16* brow[OC];
    #pragma unroll
    for (int t = 0; t < OC; ++t)
      brow[t] = sW + (size_t)((oc + t) * 16 + l15) * K + half * 8;

    for (int k0 = 0; k0 < K; k0 += 32) {
      // A fragment (16x32): half 0 -> K=k0..k0+7 & k0+16..23, half 1 -> +8
      const int ak = k0 + half * 8;
      __builtin_prefetch(arowp + ak + 32, 0, 3);      // global_prefetch next K step
      v8h alo = *(const v8h*)(arowp + ak);
      v8h ahi = *(const v8h*)(arowp + ak + 16);
      v16h a = __builtin_shufflevector(alo, ahi,
          0, 1, 2, 3, 4, 5, 6, 7, 8, 9, 10, 11, 12, 13, 14, 15);

      #pragma unroll
      for (int t = 0; t < OC; ++t) {
        // B fragment (32x16): half 0 -> K=k0..k0+7 & k0+16..23, half 1 -> +8
        v8h blo = *(const v8h*)(brow[t] + k0);
        v8h bhi = *(const v8h*)(brow[t] + k0 + 16);
        v16h b = __builtin_shufflevector(blo, bhi,
            0, 1, 2, 3, 4, 5, 6, 7, 8, 9, 10, 11, 12, 13, 14, 15);
        acc[t] = __builtin_amdgcn_wmma_f32_16x16x32_f16(
            false, a, false, b, (short)0, acc[t], false, false);
      }
    }

    #pragma unroll
    for (int t = 0; t < OC; ++t) {
      const int bcol = (oc + t) * 16 + l15;
      const float bv = bias[bcol];
      // C/D layout: VGPR r -> row = tm*16 + half*8 + r, col = bcol
      #pragma unroll
      for (int r = 0; r < 8; ++r) {
        const int row = tm * 16 + half * 8 + r;
        Y[(size_t)row * O + bcol] = acc[t][r] + bv;
      }
    }
  }
}

// ---------------------------------------------------------------------------
// Helpers: buffer init (graph-capture safe, in-stream)
// ---------------------------------------------------------------------------
__global__ void gat_fill_f32(float* __restrict__ p, float v, long long n) {
  long long i = (long long)blockIdx.x * blockDim.x + threadIdx.x;
  if (i < n) p[i] = v;
}

// ordered-int float atomic max (works with -inf init)
__device__ inline void atomicMaxF32(float* addr, float val) {
  if (val >= 0.0f)
    atomicMax((int*)addr, __float_as_int(val));
  else
    atomicMin((unsigned int*)addr, __float_as_uint(val));
}

__device__ inline void edge_endpoints(int e, int Eorig, const int* __restrict__ ei,
                                      int& s, int& d) {
  if (e < Eorig) { s = ei[e]; d = ei[Eorig + e]; }
  else           { s = d = e - Eorig; }          // appended self loops
}

// ---------------------------------------------------------------------------
// E1: score[e,h] = sum_c leakyrelu(xl[s,h,c]+xr[d,h,c]) * att[h,c]; smax = segmax
// ---------------------------------------------------------------------------
__global__ void gat_edge_score(const float* __restrict__ xl, const float* __restrict__ xr,
                               const int* __restrict__ ei, const float* __restrict__ att,
                               float* __restrict__ score, float* __restrict__ smax,
                               int Etot, int Eorig, int h, int c) {
  long long idx = (long long)blockIdx.x * blockDim.x + threadIdx.x;
  long long tot = (long long)Etot * h;
  if (idx >= tot) return;
  int e = (int)(idx / h), hh = (int)(idx % h);
  int s, d; edge_endpoints(e, Eorig, ei, s, d);

  const float4* __restrict__ xls = (const float4*)(xl + ((size_t)s * h + hh) * c);
  const float4* __restrict__ xrd = (const float4*)(xr + ((size_t)d * h + hh) * c);
  const float4* __restrict__ at  = (const float4*)(att + (size_t)hh * c);
  float sc = 0.0f;
  const int c4 = c >> 2;
  for (int j = 0; j < c4; ++j) {
    float4 l = xls[j], r = xrd[j], w = at[j];
    float m;
    m = l.x + r.x; m = m > 0.0f ? m : 0.2f * m; sc += m * w.x;
    m = l.y + r.y; m = m > 0.0f ? m : 0.2f * m; sc += m * w.y;
    m = l.z + r.z; m = m > 0.0f ? m : 0.2f * m; sc += m * w.z;
    m = l.w + r.w; m = m > 0.0f ? m : 0.2f * m; sc += m * w.w;
  }
  score[idx] = sc;
  atomicMaxF32(&smax[(size_t)d * h + hh], sc);
}

// ---------------------------------------------------------------------------
// E2: ex = exp(score - smax[dst]); den[dst] += ex
// ---------------------------------------------------------------------------
__global__ void gat_edge_exp(const int* __restrict__ ei, const float* __restrict__ score,
                             const float* __restrict__ smax, float* __restrict__ ex,
                             float* __restrict__ den, int Etot, int Eorig, int h) {
  long long idx = (long long)blockIdx.x * blockDim.x + threadIdx.x;
  long long tot = (long long)Etot * h;
  if (idx >= tot) return;
  int e = (int)(idx / h), hh = (int)(idx % h);
  int s, d; edge_endpoints(e, Eorig, ei, s, d);
  float v = expf(score[idx] - smax[(size_t)d * h + hh]);
  ex[idx] = v;
  atomicAdd(&den[(size_t)d * h + hh], v);
}

// ---------------------------------------------------------------------------
// E3: out[dst, h, c] += (ex/(den+eps)) * xl[src, h, c]   (one thread per edge-feature)
// ---------------------------------------------------------------------------
__global__ void gat_edge_scatter(const int* __restrict__ ei, const float* __restrict__ ex,
                                 const float* __restrict__ den, const float* __restrict__ xl,
                                 float* __restrict__ out, int Etot, int Eorig,
                                 int h, int c) {
  const int O = h * c;
  long long idx = (long long)blockIdx.x * blockDim.x + threadIdx.x;
  long long tot = (long long)Etot * O;
  if (idx >= tot) return;
  int e = (int)(idx / O);
  int o = (int)(idx % O);
  int hh = o / c;
  int s, d; edge_endpoints(e, Eorig, ei, s, d);
  float alpha = ex[(size_t)e * h + hh] / (den[(size_t)d * h + hh] + 1e-16f);
  atomicAdd(&out[(size_t)d * O + o], alpha * xl[(size_t)s * O + o]);
}

// ---------------------------------------------------------------------------
// E4: y = relu(out + b)
// ---------------------------------------------------------------------------
__global__ void gat_bias_relu(const float* __restrict__ out, const float* __restrict__ b,
                              float* __restrict__ y, long long n, int O) {
  long long i = (long long)blockIdx.x * blockDim.x + threadIdx.x;
  if (i >= n) return;
  float v = out[i] + b[(int)(i % O)];
  y[i] = v > 0.0f ? v : 0.0f;
}

// ---------------------------------------------------------------------------
// Host orchestration
// ---------------------------------------------------------------------------
extern "C" void kernel_launch(void* const* d_in, const int* in_sizes, int n_in,
                              void* d_out, int out_size, void* d_ws, size_t ws_size,
                              hipStream_t stream) {
  (void)n_in; (void)out_size; (void)ws_size;
  const float* x  = (const float*)d_in[0];
  const int*   ei = (const int*)d_in[1];

  const int DIN0  = 256;
  const int Nn    = in_sizes[0] / DIN0;   // 20000
  const int Eorig = in_sizes[1] / 2;      // 320000
  const int Etot  = Eorig + Nn;           // + self loops

  // workspace layout (fp32 units; every block is a multiple of 4 floats -> 16B aligned)
  float* ws = (float*)d_ws;
  size_t off = 0;
  float* xl     = ws + off; off += (size_t)Nn * 256;
  float* xr     = ws + off; off += (size_t)Nn * 256;
  float* hbuf   = ws + off; off += (size_t)Nn * 256;
  float* outbuf = ws + off; off += (size_t)Nn * 256;
  float* score  = ws + off; off += (size_t)Etot * 4;
  float* exb    = ws + off; off += (size_t)Etot * 4;
  float* smax   = ws + off; off += (size_t)Nn * 4;
  float* den    = ws + off; off += (size_t)Nn * 4;
  _Float16* X16 = (_Float16*)(ws + off); off += (size_t)Nn * 128;   // N*256 halves
  _Float16* W16 = (_Float16*)(ws + off); off += 256 * 128;          // 256*256 halves

  const int dins[4] = {256, 256, 128, 128};
  const int hs[4]   = {4, 4, 4, 1};
  const int cs[4]   = {64, 32, 32, 32};

  const float* cur = x;
  for (int L = 0; L < 4; ++L) {
    const int K = dins[L], h = hs[L], c = cs[L], O = h * c;
    const float* Wl  = (const float*)d_in[2 + 6 * L + 0];
    const float* bl  = (const float*)d_in[2 + 6 * L + 1];
    const float* Wr  = (const float*)d_in[2 + 6 * L + 2];
    const float* br  = (const float*)d_in[2 + 6 * L + 3];
    const float* att = (const float*)d_in[2 + 6 * L + 4];
    const float* bb  = (const float*)d_in[2 + 6 * L + 5];

    // convert layer input once: X16 = fp16(cur)
    const long long nk8 = (long long)Nn * K / 8;
    gat_cvt_f16<<<(unsigned)((nk8 + 255) / 256), 256, 0, stream>>>(cur, X16, nk8);

    const int gblks = (Nn / 16 + 7) / 8;            // 8 waves x 16 rows per block
    const size_t shBytes = (size_t)O * K * sizeof(_Float16);
    const long long wk8 = (long long)O * K / 8;

    // xl = cur @ Wl^T + bl
    gat_cvt_f16<<<(unsigned)((wk8 + 255) / 256), 256, 0, stream>>>(Wl, W16, wk8);
    if ((O / 16) % 4 == 0)
      gat_gemm_wmma<4><<<gblks, 256, shBytes, stream>>>(X16, W16, bl, xl, Nn, K, O);
    else
      gat_gemm_wmma<2><<<gblks, 256, shBytes, stream>>>(X16, W16, bl, xl, Nn, K, O);

    // xr = cur @ Wr^T + br
    gat_cvt_f16<<<(unsigned)((wk8 + 255) / 256), 256, 0, stream>>>(Wr, W16, wk8);
    if ((O / 16) % 4 == 0)
      gat_gemm_wmma<4><<<gblks, 256, shBytes, stream>>>(X16, W16, br, xr, Nn, K, O);
    else
      gat_gemm_wmma<2><<<gblks, 256, shBytes, stream>>>(X16, W16, br, xr, Nn, K, O);

    // init accumulators
    const long long no = (long long)Nn * O;
    const long long nh = (long long)Nn * h;
    gat_fill_f32<<<(unsigned)((no + 255) / 256), 256, 0, stream>>>(outbuf, 0.0f, no);
    gat_fill_f32<<<(unsigned)((nh + 255) / 256), 256, 0, stream>>>(den, 0.0f, nh);
    gat_fill_f32<<<(unsigned)((nh + 255) / 256), 256, 0, stream>>>(smax, -INFINITY, nh);

    // edge phase
    const long long eh = (long long)Etot * h;
    gat_edge_score<<<(unsigned)((eh + 255) / 256), 256, 0, stream>>>(
        xl, xr, ei, att, score, smax, Etot, Eorig, h, c);
    gat_edge_exp<<<(unsigned)((eh + 255) / 256), 256, 0, stream>>>(
        ei, score, smax, exb, den, Etot, Eorig, h);
    const long long eo = (long long)Etot * O;
    gat_edge_scatter<<<(unsigned)((eo + 255) / 256), 256, 0, stream>>>(
        ei, exb, den, xl, outbuf, Etot, Eorig, h, c);

    // bias + relu (all 4 layers are relu-wrapped in the reference)
    float* dst = (L == 3) ? (float*)d_out : hbuf;
    gat_bias_relu<<<(unsigned)((no + 255) / 256), 256, 0, stream>>>(outbuf, bb, dst, no, O);
    cur = hbuf;
  }
}
